// Attention_35218731827231
// MI455X (gfx1250) — matrix operand order repaired and checked by reading, not compiled, not run on hardware
//
#include <hip/hip_runtime.h>

typedef __attribute__((ext_vector_type(16))) _Float16 v16h;
typedef __attribute__((ext_vector_type(8)))  _Float16 v8h;
typedef __attribute__((ext_vector_type(8)))  float    v8f;
typedef __attribute__((ext_vector_type(4)))  int      v4i;

union V16 { v16h v; v8h h[2]; };

#define WMMA_F16(A, B, C) \
    __builtin_amdgcn_wmma_f32_16x16x32_f16(false, (A), false, (B), (short)0, (C), false, false)

// Async global->LDS (ASYNCcnt path), guarded so the file compiles on any toolchain.
// Builtin expects v4i* pointers (b128 payload), global AS1 src / LDS AS3 dst.
#if __has_builtin(__builtin_amdgcn_global_load_async_to_lds_b128) && \
    __has_builtin(__builtin_amdgcn_s_wait_asynccnt)
#define USE_ASYNC_LDS 1
typedef __attribute__((address_space(1))) v4i v4i_as1;
typedef __attribute__((address_space(3))) v4i v4i_as3;
#define ASYNC_B128(lds, g)                                                 \
    __builtin_amdgcn_global_load_async_to_lds_b128(                        \
        (v4i_as1*)(g), (v4i_as3*)(lds), 0, 0)
#define ASYNC_WAIT() __builtin_amdgcn_s_wait_asynccnt(0)
#else
#define USE_ASYNC_LDS 0
#endif

// ---------------------------------------------------------------------------
// GEMM: C[M,N](f32) = A[M,K](f32) * W[N,K]^T, M=4096, K=1024.
// Block: 256 thr = 8 waves; block tile 128(M) x 64(N); wave tile 32x32
// -> 4 v_wmma per K-step with A/B fragment reuse.
// ---------------------------------------------------------------------------
__global__ __launch_bounds__(256) void gemm_f32a(const float* __restrict__ A,
                                                 const float* __restrict__ W,
                                                 float* __restrict__ C,
                                                 int ldc, int colOff) {
    __shared__ __align__(16) _Float16 Xs[128][32];  // A tile (f16)
    __shared__ __align__(16) _Float16 Ws[32][64];   // W^T tile: Ws[k][n]
    const int t = threadIdx.x;
    const int lane = t & 31, wave = t >> 5;
    const int mw = wave >> 1, nw = wave & 1;        // 4 M-waves x 2 N-waves
    const int m0 = blockIdx.y * 128;
    const int n0 = blockIdx.x * 64;
    const int r  = lane & 15;
    const int hi = lane >> 4;
    const int ks = hi * 8;
    const int wn = t >> 2,  wk = (t & 3) * 8;       // W-stage: 64 n x 8 k per thread
    v8f c00 = {}, c01 = {}, c10 = {}, c11 = {};
    for (int k0 = 0; k0 < 1024; k0 += 32) {
        // stage X: 128x32 f32 -> f16 (2 chunks of 1 row x 8 cols per thread)
#pragma unroll
        for (int j = 0; j < 2; ++j) {
            const int cc = t + j * 256;
            const int row = cc >> 2, colc = (cc & 3) * 8;
            const float4* xp = (const float4*)&A[(size_t)(m0 + row) * 1024 + k0 + colc];
            float4 xa = xp[0], xb = xp[1];
            v8h hx;
            hx[0] = (_Float16)xa.x; hx[1] = (_Float16)xa.y;
            hx[2] = (_Float16)xa.z; hx[3] = (_Float16)xa.w;
            hx[4] = (_Float16)xb.x; hx[5] = (_Float16)xb.y;
            hx[6] = (_Float16)xb.z; hx[7] = (_Float16)xb.w;
            *(v8h*)&Xs[row][colc] = hx;
        }
        // stage W transposed: 64n x 32k f32 -> Ws[k][n] f16
        {
            const float4* wp = (const float4*)&W[(size_t)(n0 + wn) * 1024 + k0 + wk];
            float4 wa = wp[0], wb = wp[1];
            Ws[wk + 0][wn] = (_Float16)wa.x; Ws[wk + 1][wn] = (_Float16)wa.y;
            Ws[wk + 2][wn] = (_Float16)wa.z; Ws[wk + 3][wn] = (_Float16)wa.w;
            Ws[wk + 4][wn] = (_Float16)wb.x; Ws[wk + 5][wn] = (_Float16)wb.y;
            Ws[wk + 6][wn] = (_Float16)wb.z; Ws[wk + 7][wn] = (_Float16)wb.w;
        }
        if (k0 + 32 < 1024) {
            __builtin_prefetch(&A[(size_t)(m0 + (t >> 2)) * 1024 + k0 + 32 + (t & 3) * 8], 0, 1);
            __builtin_prefetch(&W[(size_t)(n0 + wn) * 1024 + k0 + 32 + wk], 0, 1);
        }
        __syncthreads();
        V16 a0, a1, b0, b1;
        a0.h[0] = *(const v8h*)&Xs[mw * 32 + r][ks];
        a0.h[1] = *(const v8h*)&Xs[mw * 32 + r][ks + 16];
        a1.h[0] = *(const v8h*)&Xs[mw * 32 + 16 + r][ks];
        a1.h[1] = *(const v8h*)&Xs[mw * 32 + 16 + r][ks + 16];
        for (int s = 0; s < 16; ++s) {
            const int kk = ks + (s & 7) + (s >> 3) * 16;
            b0.v[s] = Ws[kk][nw * 32 + r];
            b1.v[s] = Ws[kk][nw * 32 + 16 + r];
        }
        c00 = WMMA_F16(a0.v, b0.v, c00);
        c01 = WMMA_F16(a0.v, b1.v, c01);
        c10 = WMMA_F16(a1.v, b0.v, c10);
        c11 = WMMA_F16(a1.v, b1.v, c11);
        __syncthreads();
    }
    const int colA = colOff + n0 + nw * 32 + r;
    const int colB = colA + 16;
    const int row0 = m0 + mw * 32 + hi * 8;
    const int row1 = row0 + 16;
#pragma unroll
    for (int g = 0; g < 8; ++g) {
        C[(size_t)(row0 + g) * ldc + colA] = c00[g];
        C[(size_t)(row0 + g) * ldc + colB] = c01[g];
        C[(size_t)(row1 + g) * ldc + colA] = c10[g];
        C[(size_t)(row1 + g) * ldc + colB] = c11[g];
    }
}

// Same GEMM but A is already f16 (attention output) -> A tile is a pure copy:
// use GLOBAL_LOAD_ASYNC_TO_LDS_B128 when available.
__global__ __launch_bounds__(256) void gemm_f16a(const _Float16* __restrict__ A,
                                                 const float* __restrict__ W,
                                                 float* __restrict__ C,
                                                 int ldc) {
    __shared__ __align__(16) _Float16 Xs[128][32];
    __shared__ __align__(16) _Float16 Ws[32][64];
    const int t = threadIdx.x;
    const int lane = t & 31, wave = t >> 5;
    const int mw = wave >> 1, nw = wave & 1;
    const int m0 = blockIdx.y * 128;
    const int n0 = blockIdx.x * 64;
    const int r  = lane & 15;
    const int hi = lane >> 4;
    const int ks = hi * 8;
    const int wn = t >> 2, wk = (t & 3) * 8;
    v8f c00 = {}, c01 = {}, c10 = {}, c11 = {};
    for (int k0 = 0; k0 < 1024; k0 += 32) {
#pragma unroll
        for (int j = 0; j < 2; ++j) {
            const int cc = t + j * 256;
            const int row = cc >> 2, colc = (cc & 3) * 8;
#if USE_ASYNC_LDS
            ASYNC_B128(&Xs[row][colc], &A[(size_t)(m0 + row) * 1024 + k0 + colc]);
#else
            *(v8h*)&Xs[row][colc] = *(const v8h*)&A[(size_t)(m0 + row) * 1024 + k0 + colc];
#endif
        }
        {
            const float4* wp = (const float4*)&W[(size_t)(n0 + wn) * 1024 + k0 + wk];
            float4 wa = wp[0], wb = wp[1];
            Ws[wk + 0][wn] = (_Float16)wa.x; Ws[wk + 1][wn] = (_Float16)wa.y;
            Ws[wk + 2][wn] = (_Float16)wa.z; Ws[wk + 3][wn] = (_Float16)wa.w;
            Ws[wk + 4][wn] = (_Float16)wb.x; Ws[wk + 5][wn] = (_Float16)wb.y;
            Ws[wk + 6][wn] = (_Float16)wb.z; Ws[wk + 7][wn] = (_Float16)wb.w;
        }
        if (k0 + 32 < 1024) {
            __builtin_prefetch(&A[(size_t)(m0 + (t >> 2)) * 1024 + k0 + 32 + (t & 3) * 8], 0, 1);
            __builtin_prefetch(&W[(size_t)(n0 + wn) * 1024 + k0 + 32 + wk], 0, 1);
        }
#if USE_ASYNC_LDS
        ASYNC_WAIT();
#endif
        __syncthreads();
        V16 a0, a1, b0, b1;
        a0.h[0] = *(const v8h*)&Xs[mw * 32 + r][ks];
        a0.h[1] = *(const v8h*)&Xs[mw * 32 + r][ks + 16];
        a1.h[0] = *(const v8h*)&Xs[mw * 32 + 16 + r][ks];
        a1.h[1] = *(const v8h*)&Xs[mw * 32 + 16 + r][ks + 16];
        for (int s = 0; s < 16; ++s) {
            const int kk = ks + (s & 7) + (s >> 3) * 16;
            b0.v[s] = Ws[kk][nw * 32 + r];
            b1.v[s] = Ws[kk][nw * 32 + 16 + r];
        }
        c00 = WMMA_F16(a0.v, b0.v, c00);
        c01 = WMMA_F16(a0.v, b1.v, c01);
        c10 = WMMA_F16(a1.v, b0.v, c10);
        c11 = WMMA_F16(a1.v, b1.v, c11);
        __syncthreads();
    }
    const int colA = n0 + nw * 32 + r;
    const int colB = colA + 16;
    const int row0 = m0 + mw * 32 + hi * 8;
    const int row1 = row0 + 16;
#pragma unroll
    for (int g = 0; g < 8; ++g) {
        C[(size_t)(row0 + g) * ldc + colA] = c00[g];
        C[(size_t)(row0 + g) * ldc + colB] = c01[g];
        C[(size_t)(row1 + g) * ldc + colA] = c10[g];
        C[(size_t)(row1 + g) * ldc + colB] = c11[g];
    }
}

// ---------------------------------------------------------------------------
// RoPE + relayout: QKVraw[4096,1536] f32 -> head-major f16 Q/K/V.
// ---------------------------------------------------------------------------
__global__ __launch_bounds__(256) void rope_relayout(const float* __restrict__ QKVraw,
                                                     const float* __restrict__ fcos,
                                                     const float* __restrict__ fsin,
                                                     _Float16* __restrict__ Qh,
                                                     _Float16* __restrict__ Kh,
                                                     _Float16* __restrict__ Vh) {
    const int row = blockIdx.x;
    const int b = row >> 11, s = row & 2047;
    const float* src = QKVraw + (size_t)row * 1536;
    for (int idx = threadIdx.x; idx < 768; idx += 256) {
        if (idx < 512) {                                     // Q: 16 heads x 32 pairs
            int h = idx >> 5, f = idx & 31;
            float xr = src[h * 64 + 2 * f], xi = src[h * 64 + 2 * f + 1];
            float c = fcos[s * 32 + f], sn = fsin[s * 32 + f];
            size_t o = ((size_t)(b * 16 + h) * 2048 + s) * 64 + 2 * f;
            Qh[o]     = (_Float16)(xr * c - xi * sn);
            Qh[o + 1] = (_Float16)(xr * sn + xi * c);
        } else if (idx < 640) {                              // K: 4 heads x 32 pairs
            int j = idx - 512, h = j >> 5, f = j & 31;
            float xr = src[1024 + h * 64 + 2 * f], xi = src[1024 + h * 64 + 2 * f + 1];
            float c = fcos[s * 32 + f], sn = fsin[s * 32 + f];
            size_t o = ((size_t)(b * 4 + h) * 2048 + s) * 64 + 2 * f;
            Kh[o]     = (_Float16)(xr * c - xi * sn);
            Kh[o + 1] = (_Float16)(xr * sn + xi * c);
        } else {                                             // V: plain relayout
            int e = 2 * (idx - 640);
            int h = e >> 6, d = e & 63;
            size_t o = ((size_t)(b * 4 + h) * 2048 + s) * 64 + d;
            Vh[o]     = (_Float16)src[1280 + e];
            Vh[o + 1] = (_Float16)src[1280 + e + 1];
        }
    }
}

// ---------------------------------------------------------------------------
// Flash attention: block = 8 waves on one (b,h); wave = 16-row Q tile;
// 32-key blocks staged in LDS (V via async-to-LDS when available).
// ---------------------------------------------------------------------------
__global__ __launch_bounds__(256) void flash_attn(const _Float16* __restrict__ Qh,
                                                  const _Float16* __restrict__ Kh,
                                                  const _Float16* __restrict__ Vh,
                                                  _Float16* __restrict__ Attn) {
    __shared__ __align__(16) _Float16 Kt[64][32];      // K^T tile: [d][key]
    __shared__ __align__(16) _Float16 Vl[32][64];      // V tile:   [key][d]
    __shared__ __align__(16) _Float16 Pl[8][16][32];   // per-wave P relayout
    const int t = threadIdx.x, lane = t & 31, wave = t >> 5;
    const int qb = blockIdx.x, h = blockIdx.y, b = blockIdx.z;
    const int hk = h >> 2;                             // GQA: 4 Q heads per KV head
    const int qtile = qb * 128 + wave * 16;
    const int r = lane & 15, hi = lane >> 4, ks = hi * 8;
    const _Float16* Qp = Qh + ((size_t)(b * 16 + h) * 2048 + qtile) * 64;
    const _Float16* Kp = Kh + (size_t)(b * 4 + hk) * 2048 * 64;
    const _Float16* Vp = Vh + (size_t)(b * 4 + hk) * 2048 * 64;

    V16 aq0, aq1;                                      // Q fragments (d 0-31, 32-63)
    aq0.h[0] = *(const v8h*)&Qp[r * 64 + ks];
    aq0.h[1] = *(const v8h*)&Qp[r * 64 + ks + 16];
    aq1.h[0] = *(const v8h*)&Qp[r * 64 + 32 + ks];
    aq1.h[1] = *(const v8h*)&Qp[r * 64 + 32 + ks + 16];

    float m_i[8], l_i[8];
    v8f o[4];
#pragma unroll
    for (int g = 0; g < 8; ++g) { m_i[g] = -1e30f; l_i[g] = 0.f; }
#pragma unroll
    for (int dt = 0; dt < 4; ++dt) o[dt] = (v8f){};

    const int nkb = (qb + 1) * 4;
    const int kr = t >> 3, dc = (t & 7) * 8;           // K/V staging mapping
    for (int kb = 0; kb < nkb; ++kb) {
        const int key0 = kb * 32;
#if USE_ASYNC_LDS
        ASYNC_B128(&Vl[kr][dc], &Vp[(size_t)(key0 + kr) * 64 + dc]);
#else
        *(v8h*)&Vl[kr][dc] = *(const v8h*)&Vp[(size_t)(key0 + kr) * 64 + dc];
#endif
        v8h kvec = *(const v8h*)&Kp[(size_t)(key0 + kr) * 64 + dc];
#pragma unroll
        for (int j = 0; j < 8; ++j) Kt[dc + j][kr] = kvec[j];
#if USE_ASYNC_LDS
        ASYNC_WAIT();
#endif
        __syncthreads();

        if (key0 <= qtile + 15) {                      // causal: wave-uniform skip
            V16 b00, b10, b01, b11;
            for (int s = 0; s < 16; ++s) {
                const int kk = ks + (s & 7) + (s >> 3) * 16;
                b00.v[s] = Kt[kk][r];      b10.v[s] = Kt[32 + kk][r];
                b01.v[s] = Kt[kk][16 + r]; b11.v[s] = Kt[32 + kk][16 + r];
            }
            v8f zero = {};
            v8f c0 = WMMA_F16(aq0.v, b00.v, zero);  c0 = WMMA_F16(aq1.v, b10.v, c0);
            v8f c1 = WMMA_F16(aq0.v, b01.v, zero);  c1 = WMMA_F16(aq1.v, b11.v, c1);

            float s0[8], s1[8], rm[8];
#pragma unroll
            for (int g = 0; g < 8; ++g) {
                const int q = qtile + g + hi * 8;
                float a0 = c0[g] * 0.125f;             // 1/sqrt(64)
                float a1 = c1[g] * 0.125f;
                if (key0 + r > q)      a0 = -1e30f;
                if (key0 + 16 + r > q) a1 = -1e30f;
                s0[g] = a0; s1[g] = a1; rm[g] = fmaxf(a0, a1);
            }
#pragma unroll
            for (int g = 0; g < 8; ++g)
                for (int off = 1; off < 16; off <<= 1)
                    rm[g] = fmaxf(rm[g], __shfl_xor(rm[g], off, 32));
            float corr[8], rs[8];
#pragma unroll
            for (int g = 0; g < 8; ++g) {
                float mn = fmaxf(m_i[g], rm[g]);
                corr[g] = __expf(m_i[g] - mn);
                s0[g] = __expf(s0[g] - mn);
                s1[g] = __expf(s1[g] - mn);
                rs[g] = s0[g] + s1[g];
                m_i[g] = mn;
            }
#pragma unroll
            for (int g = 0; g < 8; ++g)
                for (int off = 1; off < 16; off <<= 1)
                    rs[g] += __shfl_xor(rs[g], off, 32);
#pragma unroll
            for (int g = 0; g < 8; ++g) l_i[g] = l_i[g] * corr[g] + rs[g];
#pragma unroll
            for (int dt = 0; dt < 4; ++dt)
#pragma unroll
                for (int g = 0; g < 8; ++g) o[dt][g] *= corr[g];

            // P: C-layout -> A-layout via per-wave LDS (same-wave LDS is in-order)
#pragma unroll
            for (int g = 0; g < 8; ++g) {
                int m = g + hi * 8;
                Pl[wave][m][r]      = (_Float16)s0[g];
                Pl[wave][m][r + 16] = (_Float16)s1[g];
            }
            V16 ap;
            ap.h[0] = *(const v8h*)&Pl[wave][r][ks];
            ap.h[1] = *(const v8h*)&Pl[wave][r][ks + 16];
#pragma unroll
            for (int dt = 0; dt < 4; ++dt) {
                V16 bv;
                for (int s = 0; s < 16; ++s)
                    bv.v[s] = Vl[ks + (s & 7) + (s >> 3) * 16][dt * 16 + r];
                o[dt] = WMMA_F16(ap.v, bv.v, o[dt]);
            }
        }
        __syncthreads();
    }

#pragma unroll
    for (int dt = 0; dt < 4; ++dt)
#pragma unroll
        for (int g = 0; g < 8; ++g) {
            int q = qtile + g + hi * 8;
            int d = dt * 16 + r;
            Attn[((size_t)(b * 2048) + q) * 1024 + h * 64 + d] =
                (_Float16)(o[dt][g] / l_i[g]);
        }
}

// ---------------------------------------------------------------------------
// Workspace layout (bytes):
//   QKVraw f32 [4096,1536] : 25,165,824
//   Qh f16 [2,16,2048,64]  :  8,388,608
//   Kh f16 [2, 4,2048,64]  :  2,097,152
//   Vh f16 [2, 4,2048,64]  :  2,097,152
//   Attn f16 [4096,1024]   :  8,388,608   (total ~44 MB)
// ---------------------------------------------------------------------------
extern "C" void kernel_launch(void* const* d_in, const int* in_sizes, int n_in,
                              void* d_out, int out_size, void* d_ws, size_t ws_size,
                              hipStream_t stream) {
    (void)in_sizes; (void)n_in; (void)out_size; (void)ws_size;
    const float* query = (const float*)d_in[0];
    const float* key   = (const float*)d_in[1];
    const float* value = (const float*)d_in[2];
    const float* fcos  = (const float*)d_in[3];
    const float* fsin  = (const float*)d_in[4];
    const float* wq    = (const float*)d_in[5];
    const float* wk    = (const float*)d_in[6];
    const float* wv    = (const float*)d_in[7];
    const float* wo    = (const float*)d_in[8];

    char* ws = (char*)d_ws;
    float*    QKVraw = (float*)ws;
    _Float16* Qh     = (_Float16*)(ws + 25165824);
    _Float16* Kh     = (_Float16*)(ws + 25165824 + 8388608);
    _Float16* Vh     = (_Float16*)(ws + 25165824 + 8388608 + 2097152);
    _Float16* Attn   = (_Float16*)(ws + 25165824 + 8388608 + 2097152 + 2097152);

    // QKV projections (WMMA GEMMs), concatenated into QKVraw columns
    gemm_f32a<<<dim3(16, 32), 256, 0, stream>>>(query, wq, QKVraw, 1536, 0);
    gemm_f32a<<<dim3(4,  32), 256, 0, stream>>>(key,   wk, QKVraw, 1536, 1024);
    gemm_f32a<<<dim3(4,  32), 256, 0, stream>>>(value, wv, QKVraw, 1536, 1280);

    // RoPE + f16 head-major relayout
    rope_relayout<<<4096, 256, 0, stream>>>(QKVraw, fcos, fsin, Qh, Kh, Vh);

    // Causal flash attention (WMMA)
    flash_attn<<<dim3(16, 16, 2), 256, 0, stream>>>(Qh, Kh, Vh, Attn);

    // Output projection (WMMA), f32 result
    gemm_f16a<<<dim3(16, 32), 256, 0, stream>>>(Attn, wo, (float*)d_out, 1024);
}
